// Neighbor_Attention_block_63823214019229
// MI455X (gfx1250) — compile-verified, matrix operand order checked by reading, not run on hardware
//
#include <hip/hip_runtime.h>
#include <hip/hip_bf16.h>

typedef _Float16 half_t;
typedef __attribute__((ext_vector_type(16))) _Float16 v16h;
typedef __attribute__((ext_vector_type(8)))  _Float16 v8h;
typedef __attribute__((ext_vector_type(8)))  float    v8f;

#define B_   4
#define N_   2048
#define C_   128
#define HD_  4
#define L_   2
#define FF_  512
#define BN_  (B_*N_)
#define SCALE_ 0.08838834764831845f   // 128^-0.5

enum { GF_RELU = 1, GF_ADD = 2, GF_OUT_F32 = 4, GF_OUT_F16 = 8, GF_OUT_T16 = 16, GF_BIAS = 32 };

// ---------------------------------------------------------------------------
// Fragment loader (wave32, v_wmma_f32_16x16x32_f16).
// A-fragment from a row-major 16x32 tile, and B-fragment from a transposed
// (contraction-contiguous) source share this layout:
//   lane<16 owns row (lane&15), K {0..7,16..23}; lane>=16 same row, K {8..15,24..31}
// -> two global_load_b128 per lane.
// ---------------------------------------------------------------------------
__device__ __forceinline__ v16h load_frag_k16(const half_t* S, int ld, int row_base, int k_base) {
    int lane = threadIdx.x & 31;
    int r    = lane & 15;
    int kh   = (lane >> 4) << 3;
    const half_t* p = S + (size_t)(row_base + r) * ld + k_base + kh;
    v8h lo = *(const v8h*)p;
    v8h hi = *(const v8h*)(p + 16);
    return __builtin_shufflevector(lo, hi, 0, 1, 2, 3, 4, 5, 6, 7, 8, 9, 10, 11, 12, 13, 14, 15);
}

__device__ __forceinline__ v8f wmma16(v16h a, v16h b, v8f c) {
    return __builtin_amdgcn_wmma_f32_16x16x32_f16(false, a, false, b, (short)0, c, false, false);
}

// ---------------------------------------------------------------------------
// fp32 -> f16 conversion (grid-stride)
// ---------------------------------------------------------------------------
__global__ void f32_to_f16_kernel(const float* __restrict__ in, half_t* __restrict__ out, long n) {
    long i  = (long)blockIdx.x * blockDim.x + threadIdx.x;
    long st = (long)gridDim.x * blockDim.x;
    for (; i < n; i += st) out[i] = (half_t)in[i];
}

// fp32 (K,N) row-major -> f16 (N,K) row-major (weights pre-transposed so the
// GEMM B-fragment loads are contiguous b128s)
__global__ void f32_to_f16_T_kernel(const float* __restrict__ in, half_t* __restrict__ out,
                                    int K, int Nn) {
    long total = (long)K * Nn;
    long st = (long)gridDim.x * blockDim.x;
    for (long i = (long)blockIdx.x * blockDim.x + threadIdx.x; i < total; i += st) {
        int k = (int)(i / Nn), n = (int)(i % Nn);
        out[(size_t)n * K + k] = (half_t)in[i];
    }
}

// ---------------------------------------------------------------------------
// LayerNorm over C_=128: one wave per row; optional f32 out / f16 out /
// residual add (add_to[row] += ln(x[row])).
// ---------------------------------------------------------------------------
__global__ void ln_kernel(const float* __restrict__ X, const float* __restrict__ g,
                          const float* __restrict__ bta,
                          float* out_f32, half_t* out_f16, float* add_to) {
    int row  = blockIdx.x;
    int lane = threadIdx.x;
    const float* xr = X + (size_t)row * C_;
    float v[4]; float s = 0.f;
#pragma unroll
    for (int i = 0; i < 4; ++i) { v[i] = xr[lane + 32 * i]; s += v[i]; }
#pragma unroll
    for (int off = 16; off >= 1; off >>= 1) s += __shfl_xor(s, off, 32);
    float mean = s * (1.0f / C_);
    float var = 0.f;
#pragma unroll
    for (int i = 0; i < 4; ++i) { float d = v[i] - mean; var += d * d; }
#pragma unroll
    for (int off = 16; off >= 1; off >>= 1) var += __shfl_xor(var, off, 32);
    float inv = rsqrtf(var * (1.0f / C_) + 1e-5f);
#pragma unroll
    for (int i = 0; i < 4; ++i) {
        int idx = lane + 32 * i;
        float y = (v[i] - mean) * inv * g[idx] + bta[idx];
        if (out_f32) out_f32[(size_t)row * C_ + idx] = y;
        if (out_f16) out_f16[(size_t)row * C_ + idx] = (half_t)y;
        if (add_to)  add_to[(size_t)row * C_ + idx] += y;
    }
}

// ---------------------------------------------------------------------------
// f16 WMMA GEMM, compile-time epilogue flags. One wave per 16x16 output tile.
// C = A(MxK, row-major f16) @ Bt^T  where Bt is (N,K) row-major f16.
// ---------------------------------------------------------------------------
template <int F>
__global__ void gemm_f16_kernel(const half_t* __restrict__ A, const half_t* __restrict__ Bt,
                                const float* __restrict__ bias,
                                float* Cf, half_t* Ch, half_t* Ct,
                                int K, int lda, int ldc, int ldh, int ldt) {
    int col0 = blockIdx.x * 16;
    int row0 = blockIdx.y * 16;
    v8f acc = {};
    for (int k = 0; k < K; k += 32) {
        v16h a = load_frag_k16(A, lda, row0, k);
        v16h b = load_frag_k16(Bt, K, col0, k);
        acc = wmma16(a, b, acc);
    }
    int lane = threadIdx.x & 31;
    int nn = lane & 15, hiw = lane >> 4;
    float bv = 0.f;
    if constexpr ((F & GF_BIAS) != 0) bv = bias[col0 + nn];
#pragma unroll
    for (int j = 0; j < 8; ++j) {
        float v = acc[j] + bv;
        if constexpr ((F & GF_RELU) != 0) v = fmaxf(v, 0.f);
        int m = row0 + j + 8 * hiw;
        int c = col0 + nn;
        if constexpr ((F & GF_ADD) != 0)     Cf[(size_t)m * ldc + c] += v;
        if constexpr ((F & GF_OUT_F32) != 0) Cf[(size_t)m * ldc + c] = v;
        if constexpr ((F & GF_OUT_F16) != 0) Ch[(size_t)m * ldh + c] = (half_t)v;
        if constexpr ((F & GF_OUT_T16) != 0) Ct[(size_t)c * ldt + m] = (half_t)v;
    }
}

// ---------------------------------------------------------------------------
// Attention pass 1: per (b,hd,16-query tile) compute row max m and sum-exp l
// over all keys, with adj==0 masking. One wave per tile.
// ml layout: [(b*HD+hd)*N + q][2] = {m, l}
// ---------------------------------------------------------------------------
__global__ void attn_pass1(const half_t* __restrict__ qh, const half_t* __restrict__ kh,
                           const half_t* __restrict__ adjh, float* __restrict__ ml) {
    int t = blockIdx.x;
    const int RT = N_ / 16;
    int rt = t % RT; t /= RT;
    int hd = t % HD_; int b = t / HD_;
    const half_t* Q  = qh + (size_t)(hd * B_ + b) * N_ * C_;
    const half_t* Km = kh + (size_t)(hd * B_ + b) * N_ * C_;
    const half_t* Ar = adjh + (size_t)b * N_ * N_ + (size_t)rt * 16 * N_;
    int q0 = rt * 16;
    v16h qa[4];
#pragma unroll
    for (int kk = 0; kk < 4; ++kk) qa[kk] = load_frag_k16(Q, C_, q0, kk * 32);
    int lane = threadIdx.x & 31;
    int nn = lane & 15, hiw = lane >> 4;
    float mrun[8], lrun[8];
#pragma unroll
    for (int j = 0; j < 8; ++j) { mrun[j] = -3.0e38f; lrun[j] = 0.f; }
    for (int kb = 0; kb < N_; kb += 16) {
        // prefetch next 16-key tile of K (4KB) into the cache hierarchy
        __builtin_prefetch((const char*)(Km + (size_t)(kb + 16) * C_) + lane * 128, 0, 3);
        v8f s = {};
#pragma unroll
        for (int kk = 0; kk < 4; ++kk) {
            v16h bf = load_frag_k16(Km, C_, kb, kk * 32);   // B-frag = rows of K (Q@K^T)
            s = wmma16(qa[kk], bf, s);
        }
#pragma unroll
        for (int j = 0; j < 8; ++j) {
            float av = (float)Ar[(size_t)(j + 8 * hiw) * N_ + kb + nn];
            bool masked = (av == 0.f);
            float sc = masked ? -3.0e38f : s[j] * SCALE_;
            float mx = sc;
#pragma unroll
            for (int off = 1; off < 16; off <<= 1) mx = fmaxf(mx, __shfl_xor(mx, off, 32));
            float newm = fmaxf(mrun[j], mx);
            float ex = masked ? 0.f : __expf(sc - newm);
            float sum = ex;
#pragma unroll
            for (int off = 1; off < 16; off <<= 1) sum += __shfl_xor(sum, off, 32);
            lrun[j] = lrun[j] * __expf(mrun[j] - newm) + sum;
            mrun[j] = newm;
        }
    }
    if (nn == 0) {
#pragma unroll
        for (int j = 0; j < 8; ++j) {
            size_t r = (size_t)(b * HD_ + hd) * N_ + q0 + j + 8 * hiw;
            ml[2 * r]     = mrun[j];
            ml[2 * r + 1] = lrun[j];
        }
    }
}

// ---------------------------------------------------------------------------
// Attention pass 2: recompute scores, p = exp(s-m)/l + adj, bounce 16x32 prob
// tile through LDS into A-fragment layout, accumulate P@V via WMMA.
// vt layout: [hd][b][f][key] so V B-fragments load as contiguous b128s.
// o written f16 concat-heads: ocat[b*N+n][hd*128+f].
// ---------------------------------------------------------------------------
__global__ void attn_pass2(const half_t* __restrict__ qh, const half_t* __restrict__ kh,
                           const half_t* __restrict__ vt, const half_t* __restrict__ adjh,
                           const float* __restrict__ ml, half_t* __restrict__ ocat) {
    __shared__ half_t pbuf[16 * 32];
    int t = blockIdx.x;
    const int RT = N_ / 16;
    int rt = t % RT; t /= RT;
    int hd = t % HD_; int b = t / HD_;
    const half_t* Q  = qh + (size_t)(hd * B_ + b) * N_ * C_;
    const half_t* Km = kh + (size_t)(hd * B_ + b) * N_ * C_;
    const half_t* Vt = vt + (size_t)(hd * B_ + b) * C_ * N_;
    const half_t* Ar = adjh + (size_t)b * N_ * N_ + (size_t)rt * 16 * N_;
    int q0 = rt * 16;
    int lane = threadIdx.x & 31;
    int nn = lane & 15, hiw = lane >> 4;
    v16h qa[4];
#pragma unroll
    for (int kk = 0; kk < 4; ++kk) qa[kk] = load_frag_k16(Q, C_, q0, kk * 32);
    float mrow[8], linv[8];
#pragma unroll
    for (int j = 0; j < 8; ++j) {
        size_t r = (size_t)(b * HD_ + hd) * N_ + q0 + j + 8 * hiw;
        mrow[j] = ml[2 * r];
        float l = ml[2 * r + 1];
        linv[j] = (l > 0.f) ? 1.0f / l : 0.f;
    }
    v8f acc[8] = {};
    for (int kb = 0; kb < N_; kb += 32) {
        // prefetch next K tile (8KB)
        __builtin_prefetch((const char*)(Km + (size_t)(kb + 32) * C_) + lane * 256, 0, 3);
#pragma unroll
        for (int h16 = 0; h16 < 2; ++h16) {
            int kb16 = kb + h16 * 16;
            v8f s = {};
#pragma unroll
            for (int kk = 0; kk < 4; ++kk) {
                v16h bf = load_frag_k16(Km, C_, kb16, kk * 32);
                s = wmma16(qa[kk], bf, s);
            }
#pragma unroll
            for (int j = 0; j < 8; ++j) {
                float av = (float)Ar[(size_t)(j + 8 * hiw) * N_ + kb16 + nn];
                float p = 0.f;
                if (av != 0.f) p = __expf(s[j] * SCALE_ - mrow[j]) * linv[j] + av;
                pbuf[(j + 8 * hiw) * 32 + h16 * 16 + nn] = (half_t)p;
            }
        }
        __syncthreads();
        v16h pa;
        {
            int r   = lane & 15;
            int kh2 = (lane >> 4) << 3;
            const half_t* p = &pbuf[r * 32 + kh2];
            v8h lo = *(const v8h*)p;
            v8h hi = *(const v8h*)(p + 16);
            pa = __builtin_shufflevector(lo, hi, 0, 1, 2, 3, 4, 5, 6, 7,
                                         8, 9, 10, 11, 12, 13, 14, 15);
        }
        __syncthreads();
#pragma unroll
        for (int nt = 0; nt < 8; ++nt) {
            v16h vb = load_frag_k16(Vt, N_, nt * 16, kb);   // B[k=key][n=f] = Vt[f][key]
            acc[nt] = wmma16(pa, vb, acc[nt]);
        }
    }
#pragma unroll
    for (int nt = 0; nt < 8; ++nt)
#pragma unroll
        for (int j = 0; j < 8; ++j) {
            size_t row = (size_t)b * N_ + q0 + j + 8 * hiw;
            ocat[row * (size_t)(HD_ * C_) + hd * C_ + nt * 16 + nn] = (half_t)acc[nt][j];
        }
}

// ---------------------------------------------------------------------------
extern "C" void kernel_launch(void* const* d_in, const int* in_sizes, int n_in,
                              void* d_out, int out_size, void* d_ws, size_t ws_size,
                              hipStream_t stream) {
    (void)in_sizes; (void)n_in; (void)out_size; (void)ws_size;
    const float* H       = (const float*)d_in[0];
    const float* adj     = (const float*)d_in[1];
    const float* Wq      = (const float*)d_in[2];
    const float* Wk      = (const float*)d_in[3];
    const float* Wv      = (const float*)d_in[4];
    const float* Wo      = (const float*)d_in[5];
    const float* bo      = (const float*)d_in[6];
    const float* ln_sa_g = (const float*)d_in[7];
    const float* ln_sa_b = (const float*)d_in[8];
    const float* ln1_g   = (const float*)d_in[9];
    const float* ln1_b   = (const float*)d_in[10];
    const float* ln2_g   = (const float*)d_in[11];
    const float* ln2_b   = (const float*)d_in[12];
    const float* W1      = (const float*)d_in[13];
    const float* b1      = (const float*)d_in[14];
    const float* W2      = (const float*)d_in[15];
    const float* b2      = (const float*)d_in[16];
    const float* lnf_g   = (const float*)d_in[17];
    const float* lnf_b   = (const float*)d_in[18];

    char* ws = (char*)d_ws;
    size_t off = 0;
    auto take = [&](size_t bytes) -> char* {
        char* p = ws + off;
        off = (off + bytes + 255) & ~(size_t)255;
        return p;
    };
    float*  x    = (float*) take((size_t)BN_ * C_ * 4);
    half_t* xlh  = (half_t*)take((size_t)BN_ * C_ * 2);
    half_t* qhb  = (half_t*)take((size_t)HD_ * BN_ * C_ * 2);
    half_t* khb  = (half_t*)take((size_t)HD_ * BN_ * C_ * 2);
    half_t* vtb  = (half_t*)take((size_t)HD_ * BN_ * C_ * 2);
    half_t* adjh = (half_t*)take((size_t)B_ * N_ * N_ * 2);
    half_t* ocat = (half_t*)take((size_t)BN_ * HD_ * C_ * 2);
    half_t* hidh = (half_t*)take((size_t)BN_ * FF_ * 2);
    float*  tmpf = (float*) take((size_t)BN_ * C_ * 4);
    float*  ml   = (float*) take((size_t)B_ * HD_ * N_ * 2 * 4);
    half_t* wqt  = (half_t*)take((size_t)L_ * HD_ * C_ * C_ * 2);  // transposed f16 weights
    half_t* wkt  = (half_t*)take((size_t)L_ * HD_ * C_ * C_ * 2);
    half_t* wvt  = (half_t*)take((size_t)L_ * HD_ * C_ * C_ * 2);
    half_t* wot  = (half_t*)take((size_t)L_ * FF_ * C_ * 2);
    half_t* w1t  = (half_t*)take((size_t)L_ * C_ * FF_ * 2);
    half_t* w2t  = (half_t*)take((size_t)L_ * FF_ * C_ * 2);

    // one-time conversions (weights transposed: (K,N) -> f16 (N,K))
    f32_to_f16_kernel<<<2048, 256, 0, stream>>>(adj, adjh, (long)B_ * N_ * N_);
    for (int m = 0; m < L_ * HD_; ++m) {
        f32_to_f16_T_kernel<<<64, 256, 0, stream>>>(Wq + (size_t)m * C_ * C_,
                                                    wqt + (size_t)m * C_ * C_, C_, C_);
        f32_to_f16_T_kernel<<<64, 256, 0, stream>>>(Wk + (size_t)m * C_ * C_,
                                                    wkt + (size_t)m * C_ * C_, C_, C_);
        f32_to_f16_T_kernel<<<64, 256, 0, stream>>>(Wv + (size_t)m * C_ * C_,
                                                    wvt + (size_t)m * C_ * C_, C_, C_);
    }
    for (int l = 0; l < L_; ++l) {
        f32_to_f16_T_kernel<<<64, 256, 0, stream>>>(Wo + (size_t)l * FF_ * C_,
                                                    wot + (size_t)l * FF_ * C_, FF_, C_);
        f32_to_f16_T_kernel<<<64, 256, 0, stream>>>(W1 + (size_t)l * C_ * FF_,
                                                    w1t + (size_t)l * C_ * FF_, C_, FF_);
        f32_to_f16_T_kernel<<<64, 256, 0, stream>>>(W2 + (size_t)l * FF_ * C_,
                                                    w2t + (size_t)l * FF_ * C_, FF_, C_);
    }
    hipMemcpyAsync(x, H, (size_t)BN_ * C_ * 4, hipMemcpyDeviceToDevice, stream);

    for (int l = 0; l < L_; ++l) {
        // xl = ln1(x) -> f16
        ln_kernel<<<BN_, 32, 0, stream>>>(x, ln1_g + l * C_, ln1_b + l * C_, nullptr, xlh, nullptr);

        for (int hd = 0; hd < HD_; ++hd) {
            const half_t* wq_l = wqt + ((size_t)l * HD_ + hd) * C_ * C_;
            const half_t* wk_l = wkt + ((size_t)l * HD_ + hd) * C_ * C_;
            const half_t* wv_l = wvt + ((size_t)l * HD_ + hd) * C_ * C_;
            gemm_f16_kernel<GF_OUT_F16><<<dim3(C_ / 16, BN_ / 16), 32, 0, stream>>>(
                xlh, wq_l, nullptr, nullptr, qhb + (size_t)hd * BN_ * C_, nullptr,
                C_, C_, 0, C_, 0);
            gemm_f16_kernel<GF_OUT_F16><<<dim3(C_ / 16, BN_ / 16), 32, 0, stream>>>(
                xlh, wk_l, nullptr, nullptr, khb + (size_t)hd * BN_ * C_, nullptr,
                C_, C_, 0, C_, 0);
            for (int b = 0; b < B_; ++b) {
                gemm_f16_kernel<GF_OUT_T16><<<dim3(C_ / 16, N_ / 16), 32, 0, stream>>>(
                    xlh + (size_t)b * N_ * C_, wv_l, nullptr,
                    nullptr, nullptr, vtb + (size_t)(hd * B_ + b) * C_ * N_,
                    C_, C_, 0, 0, N_);
            }
        }

        int nblk = B_ * HD_ * (N_ / 16);
        attn_pass1<<<nblk, 32, 0, stream>>>(qhb, khb, adjh, ml);
        attn_pass2<<<nblk, 32, 0, stream>>>(qhb, khb, vtb, adjh, ml, ocat);

        // sa_pre = ocat @ Wo + bo  (f32)
        gemm_f16_kernel<GF_BIAS | GF_OUT_F32><<<dim3(C_ / 16, BN_ / 16), 32, 0, stream>>>(
            ocat, wot + (size_t)l * FF_ * C_, bo + l * C_,
            tmpf, nullptr, nullptr, FF_, FF_, C_, 0, 0);
        // x += ln_sa(sa_pre)
        ln_kernel<<<BN_, 32, 0, stream>>>(tmpf, ln_sa_g + l * C_, ln_sa_b + l * C_,
                                          nullptr, nullptr, x);
        // x2 = ln2(x) -> f16
        ln_kernel<<<BN_, 32, 0, stream>>>(x, ln2_g + l * C_, ln2_b + l * C_, nullptr, xlh, nullptr);
        // hid = relu(x2 @ W1 + b1) -> f16
        gemm_f16_kernel<GF_BIAS | GF_RELU | GF_OUT_F16><<<dim3(FF_ / 16, BN_ / 16), 32, 0, stream>>>(
            xlh, w1t + (size_t)l * C_ * FF_, b1 + l * FF_,
            nullptr, hidh, nullptr, C_, C_, 0, FF_, 0);
        // x += hid @ W2 + b2
        gemm_f16_kernel<GF_BIAS | GF_ADD><<<dim3(C_ / 16, BN_ / 16), 32, 0, stream>>>(
            hidh, w2t + (size_t)l * FF_ * C_, b2 + l * C_,
            x, nullptr, nullptr, FF_, FF_, C_, 0, 0);
    }

    // out = lnf(x) as f32
    ln_kernel<<<BN_, 32, 0, stream>>>(x, lnf_g, lnf_b, (float*)d_out, nullptr, nullptr);
}
